// MambaEncoderLayer_60541859004420
// MI455X (gfx1250) — compile-verified, matrix operand order checked by reading, not run on hardware
//
#include <hip/hip_runtime.h>
#include <hip/hip_bf16.h>

// ---------------------------------------------------------------------------
// Mamba encoder layer for MI455X (gfx1250, wave32, WMMA).
// GEMMs: v_wmma_f32_16x16x32_bf16, software-pipelined fragment loads,
// no EXEC manipulation around WMMA (all tile guards are compile-time).
// ---------------------------------------------------------------------------

typedef __attribute__((ext_vector_type(16))) __bf16 v16bf;
typedef __attribute__((ext_vector_type(8)))  __bf16 bf16x8;
typedef __attribute__((ext_vector_type(8)))  float  v8f;

#define D_MODEL 1024
#define D_STATE 16
#define D_CONV  4
#define D_INNER 2048
#define DT_RANK 64
#define B_SZ    2
#define L_SEQ   1024
#define M_ROWS  (B_SZ * L_SEQ)          // 2048
#define F_DBL   (DT_RANK + 2 * D_STATE) // 96

__device__ __forceinline__ __bf16 f2bf(float f) {
  unsigned u = __builtin_bit_cast(unsigned, f);
  unsigned r = (u + 0x7FFFu + ((u >> 16) & 1u)) >> 16;   // round-to-nearest-even
  unsigned short h = (unsigned short)r;
  return __builtin_bit_cast(__bf16, h);
}

__device__ __forceinline__ float silu(float v) {
  return v / (1.0f + __expf(-v));
}

// Load one 16x32 bf16 fragment half-pair (ISA 7.12.2 layout):
// base = row/col pointer + half*8;  v0..3 <- K[k0+half*8 .. +8),
// v4..7 <- K[k0+16+half*8 .. +8).  Two b128 loads.
__device__ __forceinline__ v16bf load_frag(const __bf16* __restrict__ base, int k0) {
  bf16x8 lo = *reinterpret_cast<const bf16x8*>(base + k0);
  bf16x8 hi = *reinterpret_cast<const bf16x8*>(base + k0 + 16);
  return __builtin_shufflevector(lo, hi, 0, 1, 2, 3, 4, 5, 6, 7,
                                 8, 9, 10, 11, 12, 13, 14, 15);
}

// --------------------------- fp32 -> bf16 convert ---------------------------
__global__ void cvt_f32_bf16(const float* __restrict__ src, __bf16* __restrict__ dst, int n) {
  int i = blockIdx.x * blockDim.x + threadIdx.x;
  if (i < n) dst[i] = f2bf(src[i]);
}

// ------------------------------- LayerNorm ----------------------------------
__global__ void ln_kernel(const float* __restrict__ x, const float* __restrict__ w,
                          const float* __restrict__ b, __bf16* __restrict__ out) {
  const int row = blockIdx.x;
  const float* xr = x + (size_t)row * D_MODEL;
  __shared__ float s1[256], s2[256];
  float sum = 0.f, sq = 0.f;
  for (int i = threadIdx.x; i < D_MODEL; i += 256) {
    float v = xr[i]; sum += v; sq += v * v;
  }
  s1[threadIdx.x] = sum; s2[threadIdx.x] = sq;
  __syncthreads();
  for (int s = 128; s > 0; s >>= 1) {
    if (threadIdx.x < s) { s1[threadIdx.x] += s1[threadIdx.x + s]; s2[threadIdx.x] += s2[threadIdx.x + s]; }
    __syncthreads();
  }
  const float mu  = s1[0] * (1.0f / D_MODEL);
  const float var = s2[0] * (1.0f / D_MODEL) - mu * mu;
  const float rs  = rsqrtf(var + 1e-5f);
  for (int i = threadIdx.x; i < D_MODEL; i += 256)
    out[(size_t)row * D_MODEL + i] = f2bf((xr[i] - mu) * rs * w[i] + b[i]);
}

// ------------------------------ WMMA GEMM -----------------------------------
// C[M,N] = A[M,K] * B[N,K]^T.  Wave tile: 16 rows x NT*16 cols, 4 waves/block.
// Two-stage software pipeline (ping-pong fragments, no rotation copies).
// Epilogues: 0 plain | 1 softplus(+bias) | 2 +residual | 3 split u/z.
template <int EPI, int NT>
__global__ void gemm_bf16_wmma(const __bf16* __restrict__ A,
                               const __bf16* __restrict__ B,
                               float* __restrict__ C,
                               int M, int N, int K,
                               const float* __restrict__ aux0,
                               float* __restrict__ aux1,
                               int split) {
  const int lane = threadIdx.x & 31;
  const int wave = __builtin_amdgcn_readfirstlane(threadIdx.x >> 5);  // SGPR
  const int m0   = blockIdx.y * 16;
  const int n0   = (blockIdx.x * 4 + wave) * (NT * 16);
  if (n0 >= N) return;                       // scalar branch (wave-uniform SGPR)
  const int half = lane >> 4;
  const int l16  = lane & 15;

  const __bf16* Ab = A + (size_t)(m0 + l16) * K + half * 8;
  const __bf16* Bb[NT];
#pragma unroll
  for (int t = 0; t < NT; ++t)
    Bb[t] = B + (size_t)(n0 + t * 16 + l16) * K + half * 8;

  v8f acc[NT];
#pragma unroll
  for (int t = 0; t < NT; ++t) acc[t] = v8f{};

  // ---- two-stage pipeline: buffers (a0,b0) and (a1,b1) ----
  v16bf a0 = load_frag(Ab, 0), a1;
  v16bf b0[NT], b1[NT];
#pragma unroll
  for (int t = 0; t < NT; ++t) b0[t] = load_frag(Bb[t], 0);

  int k0 = 0;
  while (true) {
    const int k1 = k0 + 32;
    if (k1 < K) {
      __builtin_prefetch(Ab + k1 + 24, 0, 0);          // global_prefetch_b8
      a1 = load_frag(Ab, k1);
#pragma unroll
      for (int t = 0; t < NT; ++t) b1[t] = load_frag(Bb[t], k1);
    }
#pragma unroll
    for (int t = 0; t < NT; ++t)
      acc[t] = __builtin_amdgcn_wmma_f32_16x16x32_bf16(
          false, a0, false, b0[t], (short)0, acc[t], false, false);
    if (k1 >= K) break;

    const int k2 = k1 + 32;
    if (k2 < K) {
      a0 = load_frag(Ab, k2);
#pragma unroll
      for (int t = 0; t < NT; ++t) b0[t] = load_frag(Bb[t], k2);
    }
#pragma unroll
    for (int t = 0; t < NT; ++t)
      acc[t] = __builtin_amdgcn_wmma_f32_16x16x32_bf16(
          false, a1, false, b1[t], (short)0, acc[t], false, false);
    if (k2 >= K) break;
    k0 = k2;
  }

  // C/D layout: VGPR r -> row m0 + r + 8*half, col = n0 + t*16 + lane%16
#pragma unroll
  for (int t = 0; t < NT; ++t) {
    const int col = n0 + t * 16 + l16;
#pragma unroll
    for (int r = 0; r < 8; ++r) {
      const int orow = m0 + r + 8 * half;
      float val = acc[t][r];
      if constexpr (EPI == 0) {
        C[(size_t)orow * N + col] = val;
      } else if constexpr (EPI == 1) {           // delta = softplus(acc + bias)
        val += aux0[col];
        val = (val > 20.0f) ? val : __logf(1.0f + __expf(val));
        C[(size_t)orow * N + col] = val;
      } else if constexpr (EPI == 2) {           // out = acc + residual
        C[(size_t)orow * N + col] = val + aux0[(size_t)orow * N + col];
      } else {                                   // EPI == 3 : u/z split
        if (col < split) C[(size_t)orow * split + col] = val;
        else             aux1[(size_t)orow * split + (col - split)] = val;
      }
    }
  }
}

// ----------------------- causal depthwise conv + SiLU -----------------------
__global__ void conv_silu_kernel(const float* __restrict__ u0,
                                 const float* __restrict__ cw,
                                 const float* __restrict__ cb,
                                 float* __restrict__ uc,
                                 __bf16* __restrict__ ub) {
  const size_t idx = (size_t)blockIdx.x * blockDim.x + threadIdx.x;
  if (idx >= (size_t)M_ROWS * D_INNER) return;
  const int d = (int)(idx % D_INNER);
  const size_t m = idx / D_INNER;
  const int l = (int)(m % L_SEQ);
  float acc = cb[d];
#pragma unroll
  for (int j = 0; j < D_CONV; ++j) {
    const int lp = l - (D_CONV - 1) + j;
    if (lp >= 0)
      acc += cw[d * D_CONV + j] * u0[(m - (D_CONV - 1) + j) * D_INNER + d];
  }
  const float s = silu(acc);
  uc[idx] = s;
  ub[idx] = f2bf(s);
}

// ----------------------- slice dt columns of x_dbl to bf16 ------------------
__global__ void dt_slice_kernel(const float* __restrict__ xd, __bf16* __restrict__ dtb) {
  const int idx = blockIdx.x * blockDim.x + threadIdx.x;
  if (idx >= M_ROWS * DT_RANK) return;
  const int row = idx / DT_RANK, r = idx % DT_RANK;
  dtb[idx] = f2bf(xd[(size_t)row * F_DBL + r]);
}

// ----------------------------- selective scan -------------------------------
__global__ void scan_kernel(const float* __restrict__ del,
                            const float* __restrict__ uc,
                            const float* __restrict__ xd,
                            const float* __restrict__ A_log,
                            float* __restrict__ y) {
  const int idx = blockIdx.x * blockDim.x + threadIdx.x;
  if (idx >= B_SZ * D_INNER) return;
  const int b = idx / D_INNER, d = idx % D_INNER;
  float Ar[D_STATE];
#pragma unroll
  for (int n = 0; n < D_STATE; ++n) Ar[n] = -__expf(A_log[d * D_STATE + n]);
  float h[D_STATE];
#pragma unroll
  for (int n = 0; n < D_STATE; ++n) h[n] = 0.f;
  for (int l = 0; l < L_SEQ; ++l) {
    const size_t m = (size_t)b * L_SEQ + l;
    const float dt = del[m * D_INNER + d];
    const float u  = uc[m * D_INNER + d];
    float Bv[D_STATE], Cv[D_STATE];
    const float4* bp = reinterpret_cast<const float4*>(xd + m * F_DBL + DT_RANK);
    const float4* cp = reinterpret_cast<const float4*>(xd + m * F_DBL + DT_RANK + D_STATE);
#pragma unroll
    for (int q = 0; q < D_STATE / 4; ++q) {
      float4 t4 = bp[q];
      Bv[4 * q] = t4.x; Bv[4 * q + 1] = t4.y; Bv[4 * q + 2] = t4.z; Bv[4 * q + 3] = t4.w;
      float4 c4 = cp[q];
      Cv[4 * q] = c4.x; Cv[4 * q + 1] = c4.y; Cv[4 * q + 2] = c4.z; Cv[4 * q + 3] = c4.w;
    }
    const float du = dt * u;
    float yv = 0.f;
#pragma unroll
    for (int n = 0; n < D_STATE; ++n) {
      const float dA = __expf(dt * Ar[n]);
      h[n] = dA * h[n] + du * Bv[n];
      yv += h[n] * Cv[n];
    }
    y[m * D_INNER + d] = yv;
  }
}

// ----------------------- gating: (y + u*D) * silu(z) -> bf16 ----------------
__global__ void gate_kernel(const float* __restrict__ y, const float* __restrict__ uc,
                            const float* __restrict__ z, const float* __restrict__ Dp,
                            __bf16* __restrict__ yb) {
  const size_t idx = (size_t)blockIdx.x * blockDim.x + threadIdx.x;
  if (idx >= (size_t)M_ROWS * D_INNER) return;
  const int d = (int)(idx % D_INNER);
  const float v = (y[idx] + uc[idx] * Dp[d]) * silu(z[idx]);
  yb[idx] = f2bf(v);
}

// ---------------------------------------------------------------------------
extern "C" void kernel_launch(void* const* d_in, const int* in_sizes, int n_in,
                              void* d_out, int out_size, void* d_ws, size_t ws_size,
                              hipStream_t stream) {
  const float* x         = (const float*)d_in[0];
  const float* norm_w    = (const float*)d_in[1];
  const float* norm_b    = (const float*)d_in[2];
  const float* in_proj_w = (const float*)d_in[3];
  const float* conv_w    = (const float*)d_in[4];
  const float* conv_b    = (const float*)d_in[5];
  const float* x_proj_w  = (const float*)d_in[6];
  const float* dt_proj_w = (const float*)d_in[7];
  const float* dt_proj_b = (const float*)d_in[8];
  const float* A_log     = (const float*)d_in[9];
  const float* D_param   = (const float*)d_in[10];
  const float* out_proj_w= (const float*)d_in[11];
  (void)in_sizes; (void)n_in; (void)out_size; (void)ws_size;

  size_t off = 0;
  auto carve = [&](size_t bytes) -> char* {
    char* p = (char*)d_ws + off;
    off += (bytes + 255) & ~(size_t)255;
    return p;
  };
  __bf16* XN  = (__bf16*)carve((size_t)M_ROWS * D_MODEL * 2);
  __bf16* WI  = (__bf16*)carve((size_t)2 * D_INNER * D_MODEL * 2);
  float*  U0  = (float*) carve((size_t)M_ROWS * D_INNER * 4);
  float*  Z   = (float*) carve((size_t)M_ROWS * D_INNER * 4);
  float*  UC  = (float*) carve((size_t)M_ROWS * D_INNER * 4);
  __bf16* UB  = (__bf16*)carve((size_t)M_ROWS * D_INNER * 2);
  __bf16* WX  = (__bf16*)carve((size_t)F_DBL * D_INNER * 2);
  float*  XD  = (float*) carve((size_t)M_ROWS * F_DBL * 4);
  __bf16* DTB = (__bf16*)carve((size_t)M_ROWS * DT_RANK * 2);
  __bf16* WD  = (__bf16*)carve((size_t)D_INNER * DT_RANK * 2);
  float*  Y   = (float*) carve((size_t)M_ROWS * D_INNER * 4);
  float*  DEL = U0;                   // reuse: U0 dead after conv
  __bf16* YB  = WI;                   // reuse: WI dead after in_proj GEMM
  __bf16* WO  = XN;                   // reuse: XN dead after in_proj GEMM

  const dim3 blk1(256);
  auto g1 = [](size_t n) { return dim3((unsigned)((n + 255) / 256)); };

  // 1) LayerNorm -> XN (bf16)
  ln_kernel<<<dim3(M_ROWS), blk1, 0, stream>>>(x, norm_w, norm_b, XN);

  // 2) in_proj weights -> bf16
  cvt_f32_bf16<<<g1((size_t)2 * D_INNER * D_MODEL), blk1, 0, stream>>>(
      in_proj_w, WI, 2 * D_INNER * D_MODEL);

  // 3) in_proj GEMM, u/z split epilogue: [2048,1024] x [4096,1024]^T
  gemm_bf16_wmma<3, 4><<<dim3((2 * D_INNER + 255) / 256, M_ROWS / 16),
                         dim3(128), 0, stream>>>(
      XN, WI, U0, M_ROWS, 2 * D_INNER, D_MODEL, nullptr, Z, D_INNER);

  // 4) depthwise causal conv + SiLU
  conv_silu_kernel<<<g1((size_t)M_ROWS * D_INNER), blk1, 0, stream>>>(
      U0, conv_w, conv_b, UC, UB);

  // 5) x_proj: [2048,2048] x [96,2048]^T -> XD   (NT=2 -> 32 cols/wave)
  cvt_f32_bf16<<<g1((size_t)F_DBL * D_INNER), blk1, 0, stream>>>(
      x_proj_w, WX, F_DBL * D_INNER);
  gemm_bf16_wmma<0, 2><<<dim3((F_DBL + 127) / 128, M_ROWS / 16),
                         dim3(128), 0, stream>>>(
      UB, WX, XD, M_ROWS, F_DBL, D_INNER, nullptr, nullptr, 0);

  // 6) dt_proj + softplus(+bias): [2048,64] x [2048,64]^T -> DEL
  dt_slice_kernel<<<g1((size_t)M_ROWS * DT_RANK), blk1, 0, stream>>>(XD, DTB);
  cvt_f32_bf16<<<g1((size_t)D_INNER * DT_RANK), blk1, 0, stream>>>(
      dt_proj_w, WD, D_INNER * DT_RANK);
  gemm_bf16_wmma<1, 4><<<dim3((D_INNER + 255) / 256, M_ROWS / 16),
                         dim3(128), 0, stream>>>(
      DTB, WD, DEL, M_ROWS, D_INNER, DT_RANK, dt_proj_b, nullptr, 0);

  // 7) selective scan -> Y
  scan_kernel<<<g1((size_t)B_SZ * D_INNER), dim3(128), 0, stream>>>(
      DEL, UC, XD, A_log, Y);

  // 8) gating -> YB (bf16)
  gate_kernel<<<g1((size_t)M_ROWS * D_INNER), blk1, 0, stream>>>(
      Y, UC, Z, D_param, YB);

  // 9) out_proj + residual -> d_out
  cvt_f32_bf16<<<g1((size_t)D_MODEL * D_INNER), blk1, 0, stream>>>(
      out_proj_w, WO, D_MODEL * D_INNER);
  gemm_bf16_wmma<2, 4><<<dim3((D_MODEL + 255) / 256, M_ROWS / 16),
                         dim3(128), 0, stream>>>(
      YB, WO, (float*)d_out, M_ROWS, D_MODEL, D_INNER, x, nullptr, 0);
}
